// DecoderLayer_41875931136656
// MI455X (gfx1250) — matrix operand order repaired and checked by reading, not compiled, not run on hardware
//
#include <hip/hip_runtime.h>

// ---------------------------------------------------------------------------
// Types for CDNA5 WMMA
// ---------------------------------------------------------------------------
typedef __attribute__((ext_vector_type(16))) __bf16 v16bf;
typedef __attribute__((ext_vector_type(8)))  float  v8f;

struct U32x8 { uint4 lo; uint4 hi; };   // 32 bytes == sizeof(v16bf)

__device__ __forceinline__ unsigned short f2bf(float f) {
    union { float f; unsigned u; } x; x.f = f;
    unsigned u = x.u;
    u += 0x7fffu + ((u >> 16) & 1u);    // round-to-nearest-even
    return (unsigned short)(u >> 16);
}

#define TILE_LD 40   // ushort row stride for 32-wide bf16 tiles (padded, 16B aligned)

// ---------------------------------------------------------------------------
// Straight fp32 -> bf16 conversion (4 elements / thread).
// ---------------------------------------------------------------------------
__global__ void __launch_bounds__(256)
convert_bf16_kernel(const float* __restrict__ x, unsigned short* __restrict__ y, long long n)
{
    long long i = ((long long)blockIdx.x * 256 + threadIdx.x) * 4;
    if (i + 3 >= n) { for (long long j = i; j < n; ++j) y[j] = f2bf(x[j]); return; }
    const float4 v = *(const float4*)(x + i);
    unsigned p0 = (unsigned)f2bf(v.x) | ((unsigned)f2bf(v.y) << 16);
    unsigned p1 = (unsigned)f2bf(v.z) | ((unsigned)f2bf(v.w) << 16);
    *(uint2*)(y + i) = make_uint2(p0, p1);
}

// ---------------------------------------------------------------------------
// Transpose + convert: in fp32 [R,C] -> out bf16 [C,R].  32x32 tiles, 256 thr.
// ---------------------------------------------------------------------------
__global__ void __launch_bounds__(256)
transpose_bf16_kernel(const float* __restrict__ in, unsigned short* __restrict__ out,
                      int R, int C)
{
    __shared__ float tile[32][33];
    const int r0 = blockIdx.y * 32, c0 = blockIdx.x * 32;
    const int tr  = threadIdx.x >> 3;        // 0..31
    const int tc4 = (threadIdx.x & 7) * 4;   // 0..28

    const float4 v = *(const float4*)(in + (long long)(r0 + tr) * C + c0 + tc4);
    tile[tr][tc4 + 0] = v.x; tile[tr][tc4 + 1] = v.y;
    tile[tr][tc4 + 2] = v.z; tile[tr][tc4 + 3] = v.w;
    __syncthreads();

    const int cc = tr, rr4 = tc4;            // out row = original column
    unsigned short s0 = f2bf(tile[rr4 + 0][cc]);
    unsigned short s1 = f2bf(tile[rr4 + 1][cc]);
    unsigned short s2 = f2bf(tile[rr4 + 2][cc]);
    unsigned short s3 = f2bf(tile[rr4 + 3][cc]);
    unsigned p0 = (unsigned)s0 | ((unsigned)s1 << 16);
    unsigned p1 = (unsigned)s2 | ((unsigned)s3 << 16);
    *(uint2*)(out + (long long)(c0 + cc) * R + r0 + rr4) = make_uint2(p0, p1);
}

// ---------------------------------------------------------------------------
// Batched WMMA GEMM, bf16 operands, f32 accumulate:
//   C[m,n] = act( scale * sum_k A[m,k] * B(k,n) + bias[n] )
// A: bf16 row-major [M,Kd]. B: bf16; bScatterKN==0 -> stored [N,Kd] (NT, k-contig),
// bScatterKN==1 -> stored [Kd,N] (NN, transpose-scattered into LDS).
// Outputs: Cf (fp32, optional) and/or Cb (bf16, optional), same logical layout.
// Block 256 thr = 8 waves; block tile 128x64; wave tile 32x32 (4 WMMA / K-step).
// ---------------------------------------------------------------------------
__global__ void __launch_bounds__(256)
wmma_gemm_kernel(const unsigned short* __restrict__ A, long long lda, long long asb, long long ash,
                 const unsigned short* __restrict__ Bm, long long ldb, long long bsb, long long bsh,
                 int bScatterKN,
                 const float* __restrict__ bias,
                 float* __restrict__ Cf, unsigned short* __restrict__ Cb,
                 long long ldc, long long csb, long long csh,
                 int M, int N, int Kd, float scale, int act, int Hb)
{
    __shared__ unsigned short Alds[128 * TILE_LD];
    __shared__ unsigned short Blds[64 * TILE_LD];

    const int z  = blockIdx.z;
    const int bb = z / Hb, hh = z % Hb;
    A  += (long long)bb * asb + (long long)hh * ash;
    Bm += (long long)bb * bsb + (long long)hh * bsh;
    const long long cofs = (long long)bb * csb + (long long)hh * csh;

    const int m0 = blockIdx.y * 128;
    const int n0 = blockIdx.x * 64;
    const int t    = threadIdx.x;
    const int wid  = t >> 5;
    const int lane = t & 31;
    const int wm = wid & 3;       // M wave (0..3), 32 rows each
    const int wn = wid >> 2;      // N wave (0..1), 32 cols each
    const int lh = lane >> 4;     // lane half (0/1)
    const int ll = lane & 15;

    v8f acc00 = {}, acc01 = {}, acc10 = {}, acc11 = {};

    for (int k0 = 0; k0 < Kd; k0 += 32) {
        // ---- stage A tile (128 x 32 bf16): pure b128 copies, 2 chunks/thread ----
        #pragma unroll
        for (int rep = 0; rep < 2; ++rep) {
            int cid = t + rep * 256;          // 0..511
            int row = cid >> 2;               // 0..127
            int q4  = (cid & 3) << 3;         // 0,8,16,24 (ushort offset)
            const uint4 v = *(const uint4*)(A + (long long)(m0 + row) * lda + (k0 + q4));
            *(uint4*)&Alds[row * TILE_LD + q4] = v;
        }
        // ---- stage B tile -> Blds[n][k] (64 x 32 bf16) ----
        if (!bScatterKN) {
            // NT: rows already k-contiguous; 1 chunk/thread
            int row = t >> 2;                 // 0..63
            int q4  = (t & 3) << 3;
            const uint4 v = *(const uint4*)(Bm + (long long)(n0 + row) * ldb + (k0 + q4));
            *(uint4*)&Blds[row * TILE_LD + q4] = v;
        } else {
            // NN: B stored [Kd,N]; load 32 x 64, scatter-transpose
            int r  = t >> 3;                  // k row 0..31
            int j8 = (t & 7) << 3;            // n   0..56
            union { uint4 u; unsigned short s[8]; } w;
            w.u = *(const uint4*)(Bm + (long long)(k0 + r) * ldb + (n0 + j8));
            #pragma unroll
            for (int e = 0; e < 8; ++e)
                Blds[(j8 + e) * TILE_LD + r] = w.s[e];
        }
        if (k0 + 32 < Kd)   // hint next A tile (global_prefetch_b8)
            __builtin_prefetch((const void*)(A + (long long)(m0 + (t >> 1)) * lda + (k0 + 32)), 0, 1);
        __syncthreads();

        // ---- A fragments (mi = 0,16): VGPRs 0-3 = K[lh*8..+7], 4-7 = K[16+lh*8..+7]
        U32x8 a0u, a1u;
        const int ar0 = (wm * 32 + ll) * TILE_LD;
        a0u.lo = *(const uint4*)&Alds[ar0 + lh * 8];
        a0u.hi = *(const uint4*)&Alds[ar0 + 16 + lh * 8];
        const int ar1 = (wm * 32 + 16 + ll) * TILE_LD;
        a1u.lo = *(const uint4*)&Alds[ar1 + lh * 8];
        a1u.hi = *(const uint4*)&Alds[ar1 + 16 + lh * 8];
        const v16bf af0 = __builtin_bit_cast(v16bf, a0u);
        const v16bf af1 = __builtin_bit_cast(v16bf, a1u);

        // ---- B fragments (ni = 0,16): VGPRs 0-3 = K[lh*8..+7], 4-7 = K[16+lh*8..+7]
        U32x8 b0u, b1u;
        const int br0 = (wn * 32 + ll) * TILE_LD + lh * 8;
        b0u.lo = *(const uint4*)&Blds[br0];
        b0u.hi = *(const uint4*)&Blds[br0 + 16];
        const int br1 = (wn * 32 + 16 + ll) * TILE_LD + lh * 8;
        b1u.lo = *(const uint4*)&Blds[br1];
        b1u.hi = *(const uint4*)&Blds[br1 + 16];
        const v16bf bf0 = __builtin_bit_cast(v16bf, b0u);
        const v16bf bf1 = __builtin_bit_cast(v16bf, b1u);

        acc00 = __builtin_amdgcn_wmma_f32_16x16x32_bf16(false, af0, false, bf0, (short)0, acc00, false, false);
        acc01 = __builtin_amdgcn_wmma_f32_16x16x32_bf16(false, af0, false, bf1, (short)0, acc01, false, false);
        acc10 = __builtin_amdgcn_wmma_f32_16x16x32_bf16(false, af1, false, bf0, (short)0, acc10, false, false);
        acc11 = __builtin_amdgcn_wmma_f32_16x16x32_bf16(false, af1, false, bf1, (short)0, acc11, false, false);

        __syncthreads();
    }

    // ---- epilogue: D layout is m = lh*8 + i (per 16-row sub-tile), n = ll ----
    #pragma unroll
    for (int mi = 0; mi < 2; ++mi) {
        #pragma unroll
        for (int ni = 0; ni < 2; ++ni) {
            const v8f acc = mi ? (ni ? acc11 : acc10) : (ni ? acc01 : acc00);
            const int ng  = n0 + wn * 32 + ni * 16 + ll;
            const float bv = bias ? bias[ng] : 0.0f;
            #pragma unroll
            for (int i = 0; i < 8; ++i) {
                const int mg = m0 + wm * 32 + mi * 16 + lh * 8 + i;
                float v = acc[i] * scale + bv;
                if (act) v = 0.5f * v * (1.0f + erff(v * 0.70710678118654752f));
                const long long idx = cofs + (long long)mg * ldc + ng;
                if (Cf) Cf[idx] = v;
                if (Cb) Cb[idx] = f2bf(v);
            }
        }
    }
}

// ---------------------------------------------------------------------------
// Row softmax (causal optional): fp32 scores in -> bf16 probs out. S=512,128 thr.
// ---------------------------------------------------------------------------
__global__ void __launch_bounds__(128)
softmax_kernel(const float* __restrict__ s, unsigned short* __restrict__ p16,
               int T, int S, int causal)
{
    __shared__ float red[128];
    const long long row = blockIdx.x;
    const int tq  = (int)(row % T);
    const int lim = causal ? tq : (S - 1);
    const float* p = s + row * (long long)S;
    unsigned short* q = p16 + row * (long long)S;
    const int t = threadIdx.x;

    float vals[4];
    float m = -3.0e38f;
    #pragma unroll
    for (int i = 0; i < 4; ++i) {
        int c = t + i * 128;
        float v = p[c];
        if (c > lim) v = -3.0e38f;
        vals[i] = v;
        m = fmaxf(m, v);
    }
    red[t] = m; __syncthreads();
    for (int off = 64; off > 0; off >>= 1) {
        if (t < off) red[t] = fmaxf(red[t], red[t + off]);
        __syncthreads();
    }
    m = red[0]; __syncthreads();

    float sum = 0.0f;
    #pragma unroll
    for (int i = 0; i < 4; ++i) {
        int c = t + i * 128;
        float e = (c <= lim) ? __expf(vals[i] - m) : 0.0f;
        vals[i] = e; sum += e;
    }
    red[t] = sum; __syncthreads();
    for (int off = 64; off > 0; off >>= 1) {
        if (t < off) red[t] += red[t + off];
        __syncthreads();
    }
    const float inv = 1.0f / red[0];
    #pragma unroll
    for (int i = 0; i < 4; ++i) q[t + i * 128] = f2bf(vals[i] * inv);
}

// ---------------------------------------------------------------------------
// y = LayerNorm(x + a) * g + b, eps=1e-3; optional bf16 copy yb. D=512, 128 thr.
// ---------------------------------------------------------------------------
__global__ void __launch_bounds__(128)
add_ln_kernel(const float* __restrict__ x, const float* __restrict__ a,
              const float* __restrict__ g, const float* __restrict__ b,
              float* __restrict__ y, unsigned short* __restrict__ yb, int D)
{
    __shared__ float red[128];
    const long long row = blockIdx.x;
    const float* px = x + row * D;
    const float* pa = a + row * D;
    const int t = threadIdx.x;

    float v[4]; float s = 0.0f;
    #pragma unroll
    for (int i = 0; i < 4; ++i) { int c = t + i * 128; float u = px[c] + pa[c]; v[i] = u; s += u; }
    red[t] = s; __syncthreads();
    for (int off = 64; off > 0; off >>= 1) { if (t < off) red[t] += red[t + off]; __syncthreads(); }
    const float mean = red[0] / (float)D; __syncthreads();

    float q = 0.0f;
    #pragma unroll
    for (int i = 0; i < 4; ++i) { float d = v[i] - mean; q += d * d; }
    red[t] = q; __syncthreads();
    for (int off = 64; off > 0; off >>= 1) { if (t < off) red[t] += red[t + off]; __syncthreads(); }
    const float rstd = rsqrtf(red[0] / (float)D + 1e-3f);
    #pragma unroll
    for (int i = 0; i < 4; ++i) {
        int c = t + i * 128;
        float o = (v[i] - mean) * rstd * g[c] + b[c];
        y[row * D + c] = o;
        if (yb) yb[row * D + c] = f2bf(o);
    }
}

// ---------------------------------------------------------------------------
// Host-side helpers
// ---------------------------------------------------------------------------
static inline void launch_gemm(hipStream_t stream,
    const unsigned short* A, long long lda, long long asb, long long ash,
    const unsigned short* B, long long ldb, long long bsb, long long bsh, int bScatterKN,
    const float* bias, float* Cf, unsigned short* Cb,
    long long ldc, long long csb, long long csh,
    int M, int N, int Kd, float scale, int act, int batches, int Hb)
{
    dim3 grid(N / 64, M / 128, batches);
    wmma_gemm_kernel<<<grid, dim3(256), 0, stream>>>(A, lda, asb, ash, B, ldb, bsb, bsh,
                                                     bScatterKN, bias, Cf, Cb, ldc, csb, csh,
                                                     M, N, Kd, scale, act, Hb);
}

static inline void cvt(hipStream_t stream, const float* x, unsigned short* y, long long n) {
    convert_bf16_kernel<<<dim3((unsigned)((n + 1023) / 1024)), dim3(256), 0, stream>>>(x, y, n);
}
static inline void tcvt(hipStream_t stream, const float* in, unsigned short* out, int R, int C) {
    transpose_bf16_kernel<<<dim3(C / 32, R / 32), dim3(256), 0, stream>>>(in, out, R, C);
}

static inline void run_mha(hipStream_t stream,
    const unsigned short* q_in_b, const unsigned short* kv_in_b,
    const unsigned short* wqt, const float* bq, const unsigned short* wkt, const float* bk,
    const unsigned short* wvt, const float* bv, const unsigned short* wot, const float* bo,
    unsigned short* qb, unsigned short* kb, unsigned short* vb,
    float* sc, unsigned short* pr, float* pb, int causal)
{
    const int Bc = 8, Tn = 512, Sn = 512, Dd = 512, Hh = 8, Kk = 512, HK = 4096, M = 4096;
    const float iscale = 0.04419417382415922f;   // 1/sqrt(512)

    // Q/K/V projections: [4096,512] x [512,4096] + bias -> bf16
    launch_gemm(stream, q_in_b,  Dd, 0, 0, wqt, Dd, 0, 0, 0, bq, nullptr, qb, HK, 0, 0, M, HK, Dd, 1.0f, 0, 1, 1);
    launch_gemm(stream, kv_in_b, Dd, 0, 0, wkt, Dd, 0, 0, 0, bk, nullptr, kb, HK, 0, 0, M, HK, Dd, 1.0f, 0, 1, 1);
    launch_gemm(stream, kv_in_b, Dd, 0, 0, wvt, Dd, 0, 0, 0, bv, nullptr, vb, HK, 0, 0, M, HK, Dd, 1.0f, 0, 1, 1);

    // scores[b,h,t,s] = (Q . K^T)/sqrt(K): batched over (b,h)=64, both NT
    launch_gemm(stream, qb, HK, (long long)Tn * HK, Kk,
                        kb, HK, (long long)Sn * HK, Kk, 0, nullptr,
                        sc, nullptr, Sn, (long long)Hh * Tn * Sn, (long long)Tn * Sn,
                        Tn, Sn, Kk, iscale, 0, Bc * Hh, Hh);

    softmax_kernel<<<dim3(Bc * Hh * Tn), dim3(128), 0, stream>>>(sc, pr, Tn, Sn, causal);

    // O[b,t,h,k] = probs . V (V stored [S,HK] -> scatter staging); bf16 into qb
    launch_gemm(stream, pr, Sn, (long long)Hh * Tn * Sn, (long long)Tn * Sn,
                        vb, HK, (long long)Sn * HK, Kk, 1, nullptr,
                        nullptr, qb, HK, (long long)Tn * HK, Kk,
                        Tn, Kk, Sn, 1.0f, 0, Bc * Hh, Hh);

    // output projection: [4096,4096] x [4096,512] + bias -> fp32
    launch_gemm(stream, qb, HK, 0, 0, wot, HK, 0, 0, 0, bo, pb, nullptr, Dd, 0, 0, M, Dd, HK, 1.0f, 0, 1, 1);
}

extern "C" void kernel_launch(void* const* d_in, const int* in_sizes, int n_in,
                              void* d_out, int out_size, void* d_ws, size_t ws_size,
                              hipStream_t stream)
{
    (void)in_sizes; (void)n_in; (void)out_size; (void)ws_size;
    typedef unsigned short us;

    const float* in_seq  = (const float*)d_in[0];
    const float* out_seq = (const float*)d_in[1];
    const float* sa_wq = (const float*)d_in[2];  const float* sa_bq = (const float*)d_in[3];
    const float* sa_wk = (const float*)d_in[4];  const float* sa_bk = (const float*)d_in[5];
    const float* sa_wv = (const float*)d_in[6];  const float* sa_bv = (const float*)d_in[7];
    const float* sa_wo = (const float*)d_in[8];  const float* sa_bo = (const float*)d_in[9];
    const float* sa_lng = (const float*)d_in[10]; const float* sa_lnb = (const float*)d_in[11];
    const float* ca_wq = (const float*)d_in[12]; const float* ca_bq = (const float*)d_in[13];
    const float* ca_wk = (const float*)d_in[14]; const float* ca_bk = (const float*)d_in[15];
    const float* ca_wv = (const float*)d_in[16]; const float* ca_bv = (const float*)d_in[17];
    const float* ca_wo = (const float*)d_in[18]; const float* ca_bo = (const float*)d_in[19];
    const float* ca_lng = (const float*)d_in[20]; const float* ca_lnb = (const float*)d_in[21];
    const float* ff_w1 = (const float*)d_in[22]; const float* ff_b1 = (const float*)d_in[23];
    const float* ff_w2 = (const float*)d_in[24]; const float* ff_b2 = (const float*)d_in[25];
    const float* ff_lng = (const float*)d_in[26]; const float* ff_lnb = (const float*)d_in[27];

    // ---- workspace carve-up ----
    const long long NTOK = 4096;                 // B*T = B*S
    const long long DD = 512, HK = 4096;
    const long long SMALL = NTOK * DD;           // 2M elements
    const long long BIG   = NTOK * HK;           // 16.7M elements

    char* p = (char*)d_ws;
    auto alloc = [&](long long bytes) { char* r = p; p += (bytes + 255) & ~255LL; return (void*)r; };

    us* xi_b = (us*)alloc(SMALL * 2);            // bf16(in_seq)
    us* xo_b = (us*)alloc(SMALL * 2);            // bf16(out_seq)
    us* sa_wqt = (us*)alloc(SMALL * 4 * 2);      // [HK,512] bf16 transposed weights
    us* sa_wkt = (us*)alloc(SMALL * 4 * 2);
    us* sa_wvt = (us*)alloc(SMALL * 4 * 2);
    us* sa_wot = (us*)alloc(SMALL * 4 * 2);      // [512,HK]
    us* ca_wqt = (us*)alloc(SMALL * 4 * 2);
    us* ca_wkt = (us*)alloc(SMALL * 4 * 2);
    us* ca_wvt = (us*)alloc(SMALL * 4 * 2);
    us* ca_wot = (us*)alloc(SMALL * 4 * 2);
    us* ff_w1t = (us*)alloc(DD * 1024 * 2);      // [1024,512]
    us* ff_w2t = (us*)alloc(DD * 1024 * 2);      // [512,1024]
    us* qb = (us*)alloc(BIG * 2);                // Q bf16 / attn-out (aliased)
    us* kb = (us*)alloc(BIG * 2);                // K bf16
    us* vb = (us*)alloc(BIG * 2);                // V bf16
    us* pr = (us*)alloc(BIG * 2);                // probs bf16
    float* sc = (float*)alloc(BIG * 4);          // scores fp32
    float* pj = (float*)alloc(SMALL * 4);        // projection output fp32
    float* x1 = (float*)alloc(SMALL * 4);
    float* x2 = (float*)alloc(SMALL * 4);
    us* x1b = (us*)alloc(SMALL * 2);
    us* x2b = (us*)alloc(SMALL * 2);
    us* hb  = (us*)alloc(NTOK * 1024 * 2);       // FFN hidden bf16
    float* y = (float*)d_out;

    // ---- one-shot conversions ----
    cvt(stream, in_seq,  xi_b, SMALL);
    cvt(stream, out_seq, xo_b, SMALL);
    tcvt(stream, sa_wq, sa_wqt, 512, 4096);
    tcvt(stream, sa_wk, sa_wkt, 512, 4096);
    tcvt(stream, sa_wv, sa_wvt, 512, 4096);
    tcvt(stream, sa_wo, sa_wot, 4096, 512);
    tcvt(stream, ca_wq, ca_wqt, 512, 4096);
    tcvt(stream, ca_wk, ca_wkt, 512, 4096);
    tcvt(stream, ca_wv, ca_wvt, 512, 4096);
    tcvt(stream, ca_wo, ca_wot, 4096, 512);
    tcvt(stream, ff_w1, ff_w1t, 512, 1024);
    tcvt(stream, ff_w2, ff_w2t, 1024, 512);

    // ---- causal self-attention + add&LN ----
    run_mha(stream, xo_b, xo_b,
            sa_wqt, sa_bq, sa_wkt, sa_bk, sa_wvt, sa_bv, sa_wot, sa_bo,
            qb, kb, vb, sc, pr, pj, /*causal=*/1);
    add_ln_kernel<<<dim3(4096), dim3(128), 0, stream>>>(out_seq, pj, sa_lng, sa_lnb, x1, x1b, 512);

    // ---- cross-attention + add&LN ----
    run_mha(stream, x1b, xi_b,
            ca_wqt, ca_bq, ca_wkt, ca_bk, ca_wvt, ca_bv, ca_wot, ca_bo,
            qb, kb, vb, sc, pr, pj, /*causal=*/0);
    add_ln_kernel<<<dim3(4096), dim3(128), 0, stream>>>(x1, pj, ca_lng, ca_lnb, x2, x2b, 512);

    // ---- FFN: gelu(x2 @ w1 + b1) @ w2 + b2, add&LN ----
    launch_gemm(stream, x2b, 512, 0, 0, ff_w1t, 512, 0, 0, 0, ff_b1, nullptr, hb,
                1024, 0, 0, 4096, 1024, 512, 1.0f, /*gelu=*/1, 1, 1);
    launch_gemm(stream, hb, 1024, 0, 0, ff_w2t, 1024, 0, 0, 0, ff_b2, pj, nullptr,
                512, 0, 0, 4096, 512, 1024, 1.0f, 0, 1, 1);
    add_ln_kernel<<<dim3(4096), dim3(128), 0, stream>>>(x2, pj, ff_lng, ff_lnb, y, nullptr, 512);
}
